// FuzzySphere_16681652977959
// MI455X (gfx1250) — matrix operand order repaired and checked by reading, not compiled, not run on hardware
//
#include <hip/hip_runtime.h>
#include <hip/hip_bf16.h>
#include <math.h>

// CDNA5 / gfx1250, wave32. One wave handles one (b,m) query point.
// out[b,m,c] = sum_k NF[k,c] * sum_j S[k,j] * W[j,c]
//            = sum_j W[j,c] * (S^T x NF)[j,c]   -> v_wmma_f32_16x16x32_f16
// Filter weights (4KB) staged into LDS by the Tensor Data Mover (TDM).

typedef __attribute__((ext_vector_type(16))) _Float16 v16h;
typedef __attribute__((ext_vector_type(8)))  float    v8f;
typedef __attribute__((ext_vector_type(4)))  unsigned int tdm_v4u;
typedef __attribute__((ext_vector_type(8)))  int          tdm_v8i;
typedef __attribute__((ext_vector_type(4)))  int          tdm_v4i;

#define B_   4
#define N_   8192
#define M_   2048
#define K_   32
#define C_   64
#define NB_  16     // N_AZ*N_EL*N_RAD bins
#define WPB  4      // waves per block

__global__ __launch_bounds__(128) void fuzzy_sphere_wmma_kernel(
    const float*     __restrict__ database,   // (B,N,3)
    const float*     __restrict__ query,      // (B,M,3)
    const long long* __restrict__ nn_index,   // (B,M,K) int64
    const float*     __restrict__ nn_dist,    // (B,M,K)
    const float*     __restrict__ feats,      // (B,N,C)
    const float*     __restrict__ fw,         // (16,C,1)
    float*           __restrict__ out)        // (B,M,C)
{
    __shared__ float    Wl[NB_ * C_];          // filter weights (TDM-loaded)
    __shared__ float    Sl[WPB][K_][NB_];      // per-wave scattered coeffs (f32)
    __shared__ _Float16 NFl[WPB][K_][C_];      // per-wave gathered features (f16)

    const int tid  = threadIdx.x;
    const int wave = tid >> 5;
    const int lane = tid & 31;

    // ---- TDM: async DMA filter_weights (16x64 f32 = 4KB, contiguous) -> LDS.
    // D# group0: count=1, lds_addr, global_addr[120:64], type=2.
    // D# group1: data_size=2 (4B), tensor_dim0=tile_dim0=1024, tile_dim1=1,
    //            tensor_dim0_stride=1024. Groups 2/3 zero (<=2D tensor).
    if (wave == 0) {
        const unsigned int lds_off = (unsigned int)(uintptr_t)(void*)Wl;
        const unsigned long long ga = (unsigned long long)(uintptr_t)fw;
        tdm_v4u g0;
        g0[0] = 1u;                                   // count=1, is_restore=0
        g0[1] = lds_off;                              // lds_addr (bytes)
        g0[2] = (unsigned int)(ga & 0xFFFFFFFFull);   // global_addr[95:64]
        g0[3] = (unsigned int)((ga >> 32) & 0x01FFFFFFull) | (2u << 30); // +type=2
        tdm_v8i g1;
        g1[0] = (int)(2u << 16);                      // data_size=2 -> 4 bytes
        g1[1] = (int)(1024u << 16);                   // tensor_dim0 low16 @ [63:48]
        g1[2] = (int)(1u << 16);                      // tensor_dim1=1 low16 @ [95:80]
        g1[3] = (int)(1024u << 16);                   // tile_dim0=1024 @ [127:112]
        g1[4] = 1;                                    // tile_dim1=1 @ [143:128]
        g1[5] = 1024;                                 // tensor_dim0_stride low32
        g1[6] = (int)(1024u << 16);                   // tensor_dim1_stride low16
        g1[7] = 0;
        tdm_v4i gz4 = {0, 0, 0, 0};
        tdm_v8i gz8 = {0, 0, 0, 0, 0, 0, 0, 0};
        // 6-arg form (clang-23 / therock-10.0): (g0, g1, g2, g3, g?, cpol)
        __builtin_amdgcn_tensor_load_to_lds(g0, g1, gz4, gz4, gz8, 0);
    }

    const int mg = blockIdx.x * WPB + wave;    // flat (b,m), grid sized exactly
    const int b  = mg / M_;

    // ---- Phase 1: per-lane neighbor k = lane: bins, coeffs, feature gather ----
    const long long mk  = (long long)mg * K_ + lane;
    const int   idx  = (int)nn_index[mk];      // 0 <= idx < N
    const float dist = nn_dist[mk];

    // Kick the L2 on the feature row this lane will gather (global_prefetch_b8),
    // overlapping gather latency with the transcendental bin math below.
    const float* frow = feats + ((size_t)b * N_ + (size_t)idx) * C_;
    __builtin_prefetch(frow, 0, 0);

    const float* qp = query    + (size_t)mg * 3;
    const float* dp = database + ((size_t)b * N_ + (size_t)idx) * 3;
    const float x = dp[0] - qp[0];
    const float y = dp[1] - qp[1];
    const float z = dp[2] - qp[2];

    const float PI = 3.14159265358979323846f;
    const float az = atan2f(y, x) + PI;
    float ct = z / (dist + 1e-8f);
    ct = fminf(1.0f, fmaxf(-1.0f, ct));
    const float el = acosf(ct);

    const float a_bin = az * (4.0f / (2.0f * PI));     // AZ_SCALE
    const float e_bin = el * (2.0f / PI);              // EL_SCALE
    const float r_bin = fminf(fmaxf(dist / 0.05f, 0.0f), 2.0f - 1e-6f);

    const float a_fl = floorf(a_bin), e_fl = floorf(e_bin), r_fl = floorf(r_bin);
    const float af = a_bin - a_fl, ef = e_bin - e_fl, rf = r_bin - r_fl;
    const float ai = 1.0f - af,    ei = 1.0f - ef,    ri = 1.0f - rf;

    const int a_i = (int)a_fl;
    const int e_i = (int)e_fl;
    const int r_i = (int)r_fl;
    const int e_c  = min(max(e_i,     0), 1);
    const int e_c1 = min(max(e_i + 1, 0), 1);
    const int r_c  = min(max(r_i,     0), 1);
    const int r_c1 = min(max(r_i + 1, 0), 1);
    const int ba   = (a_i & 3) * 2;
    const int ba1  = ((a_i + 1) & 3) * 2;

    const float coeff[8] = { ai*ei*ri, af*ei*ri, ai*ef*ri, af*ef*ri,
                             ai*ei*rf, af*ei*rf, ai*ef*rf, af*ef*rf };
    const int bins[8] = {
        (ba  + e_c )*2 + r_c,  (ba1 + e_c )*2 + r_c,
        (ba  + e_c1)*2 + r_c,  (ba1 + e_c1)*2 + r_c,
        (ba  + e_c )*2 + r_c1, (ba1 + e_c )*2 + r_c1,
        (ba  + e_c1)*2 + r_c1, (ba1 + e_c1)*2 + r_c1 };

    // Scatter-accumulate 8 corner coeffs into 16 bins (own LDS row, f32 exact).
    #pragma unroll
    for (int j = 0; j < NB_; ++j) Sl[wave][lane][j] = 0.0f;
    #pragma unroll
    for (int e = 0; e < 8; ++e) Sl[wave][lane][bins[e]] += coeff[e];

    // Gather this neighbor's feature row, convert f32 -> f16 into LDS.
    #pragma unroll
    for (int c0 = 0; c0 < C_; c0 += 4) {
        float4 f = *(const float4*)(frow + c0);
        NFl[wave][lane][c0 + 0] = (_Float16)f.x;
        NFl[wave][lane][c0 + 1] = (_Float16)f.y;
        NFl[wave][lane][c0 + 2] = (_Float16)f.z;
        NFl[wave][lane][c0 + 3] = (_Float16)f.w;
    }

    // TDM completion (wave 0 only), then block-wide visibility.
    if (wave == 0) __builtin_amdgcn_s_wait_tensorcnt(0);
    __syncthreads();

    // ---- Phase 2: G = S^T (16x32) x NF (32x64) via 4x v_wmma_f32_16x16x32_f16 ----
    // A layout (16-bit 16x32): lane L row M=L&15; halfs 0..7 -> K=kb+0..7,
    // halfs 8..15 -> K=kb+16..23, kb = (L<16)?0:8.
    v16h a;
    {
        const int j  = lane & 15;
        const int kb = (lane < 16) ? 0 : 8;
        #pragma unroll
        for (int h = 0; h < 16; ++h) {
            const int k = kb + ((h < 8) ? h : (h + 8));
            a[h] = (_Float16)Sl[wave][k][j];
        }
    }

    // B layout (16-bit 32x16): lane L column N=L&15; halfs 0..7 -> K=kb2+0..7,
    // halfs 8..15 -> K=kb2+16..23, kb2 = (L<16)?0:8.
    const int n   = lane & 15;
    const int kb2 = (lane < 16) ? 0 : 8;
    const int jb  = (lane < 16) ? 0 : 8;   // D rows held by this lane

    #pragma unroll
    for (int t = 0; t < 4; ++t) {
        const int c = t * 16 + n;
        v16h bm;
        #pragma unroll
        for (int h = 0; h < 16; ++h) bm[h] = NFl[wave][kb2 + ((h < 8) ? h : (h + 8))][c];

        v8f acc = {};
        // (neg_a, A, neg_b, B, c_mod, C, reuse_a, reuse_b)
        acc = __builtin_amdgcn_wmma_f32_16x16x32_f16(
                  false, a, false, bm, (short)0, acc, false, false);

        // out[c] = sum_j W[j,c] * G[j,c]; each lane holds 8 of 16 j's.
        float p = 0.0f;
        #pragma unroll
        for (int r = 0; r < 8; ++r) p += Wl[(jb + r) * C_ + c] * acc[r];
        const float tot = p + __shfl_xor(p, 16);
        if (lane < 16) out[(size_t)mg * C_ + c] = tot;
    }
}

extern "C" void kernel_launch(void* const* d_in, const int* in_sizes, int n_in,
                              void* d_out, int out_size, void* d_ws, size_t ws_size,
                              hipStream_t stream) {
    (void)in_sizes; (void)n_in; (void)out_size; (void)d_ws; (void)ws_size;
    const float*     database = (const float*)d_in[0];
    const float*     query    = (const float*)d_in[1];
    const long long* nn_index = (const long long*)d_in[2];
    // d_in[3] = nn_count: unused by the reference einsum
    const float*     nn_dist  = (const float*)d_in[4];
    const float*     feats    = (const float*)d_in[5];
    const float*     fw       = (const float*)d_in[6];
    float*           out      = (float*)d_out;

    dim3 block(128);                         // 4 wave32 waves
    dim3 grid((B_ * M_) / WPB);              // 8192 points / 4 per block = 2048
    fuzzy_sphere_wmma_kernel<<<grid, block, 0, stream>>>(
        database, query, nn_index, nn_dist, feats, fw, out);
}
